// GroupedQueryAttention_15444702396869
// MI455X (gfx1250) — matrix operand order repaired and checked by reading, not compiled, not run on hardware
//
#include <hip/hip_runtime.h>

// ---------------------------------------------------------------------------
// Types for CDNA5 WMMA (wave32): v_wmma_f32_16x16x32_bf16
// ---------------------------------------------------------------------------
typedef __attribute__((ext_vector_type(16))) __bf16         v16bf;
typedef __attribute__((ext_vector_type(8)))  float          v8f;
typedef __attribute__((ext_vector_type(4)))  unsigned int   u32x4;
typedef __attribute__((ext_vector_type(4)))  unsigned short u16x4;
typedef __attribute__((ext_vector_type(4)))  float          f32x4;
typedef __attribute__((ext_vector_type(8)))  int            i32x8;
typedef __attribute__((ext_vector_type(4)))  int            i32x4;

union Frag {            // 32 bytes: one A- or B-operand fragment per lane
    v16bf v;
    u32x4 q[2];
};

__device__ __forceinline__ unsigned short f2bf(float f) {
    unsigned u = __float_as_uint(f);
    unsigned r = u + 0x7FFFu + ((u >> 16) & 1u);   // round-to-nearest-even
    return (unsigned short)(r >> 16);
}
__device__ __forceinline__ float bf2f(unsigned short h) {
    return __uint_as_float(((unsigned)h) << 16);
}

__device__ __forceinline__ v8f wmma_bf16(const Frag& a, const Frag& b, v8f c) {
    return __builtin_amdgcn_wmma_f32_16x16x32_bf16(
        /*neg_a=*/false, a.v, /*neg_b=*/false, b.v,
        /*c_mod=*/(short)0, c, /*reuse_a=*/false, /*reuse_b=*/false);
}

__device__ __forceinline__ float red_max16(float v) {   // within 16-lane half
    v = fmaxf(v, __shfl_xor(v, 1));
    v = fmaxf(v, __shfl_xor(v, 2));
    v = fmaxf(v, __shfl_xor(v, 4));
    v = fmaxf(v, __shfl_xor(v, 8));
    return v;
}
__device__ __forceinline__ float red_sum16(float v) {
    v += __shfl_xor(v, 1);
    v += __shfl_xor(v, 2);
    v += __shfl_xor(v, 4);
    v += __shfl_xor(v, 8);
    return v;
}
__device__ __forceinline__ float red_sum32(float v) {
    v = red_sum16(v);
    v += __shfl_xor(v, 16);
    return v;
}

// ---------------------------------------------------------------------------
// Problem constants
// ---------------------------------------------------------------------------
#define BATCH   2
#define SEQ     2048
#define DMODEL  2048
#define NH      16
#define NKV     4
#define DH      128
#define ROWS    (BATCH * SEQ)           // 4096
#define QKV_N   (DMODEL + 2 * NKV * DH) // 3072

// LDS tile pitch for TDM-staged K-slices: 32 data ushorts + 4 DWORD pad = 80 B.
// Row n starts at bank (20n mod 64): all distinct multiples of 4 for n=0..15,
// so b128 reads across a 16-lane half are bank-conflict-free.
#define TPITCH  40

// ---------------------------------------------------------------------------
// Tensor Data Mover: 2D tile (tile_k elems of 2B per row, tile_rows rows) from
// row-major global (row stride = stride_elems) into LDS at byte offset
// lds_off, padding 16 B after every 64 B stored (pad_interval=3, pad_amount=3).
// D# packing per CDNA5 ISA. 6-arg builtin form
// (clang-23 / therock-10.0 headers).
// ---------------------------------------------------------------------------
__device__ __forceinline__ void tdm_load_2d(unsigned lds_off,
                                            const unsigned short* gptr,
                                            unsigned tile_k, unsigned tile_rows,
                                            unsigned stride_elems) {
    unsigned long long ga = (unsigned long long)gptr;
    u32x4 g0;
    g0[0] = 1u;                                   // count=1, user descriptor
    g0[1] = lds_off;                              // lds_addr (bytes)
    g0[2] = (unsigned)ga;                         // global_addr[31:0]
    g0[3] = (unsigned)(ga >> 32) | (2u << 30);    // global_addr[56:32] | type=2

    const unsigned dim0 = stride_elems;           // tensor_dim0 (elems)
    const unsigned dim1 = 1u << 20;               // tensor_dim1: large (no OOB)
    i32x8 g1;
    g1[0] = (int)((1u << 16) |                    // data_size = 1 -> 2 bytes
                  (1u << 20) |                    // pad_enable
                  (3u << 22) |                    // pad_interval: 16 DWORDs
                  (3u << 25));                    // pad_amount:   4 DWORDs
    g1[1] = (int)((dim0 & 0xFFFFu) << 16);                          // dim0 lo16
    g1[2] = (int)(((dim0 >> 16) & 0xFFFFu) | ((dim1 & 0xFFFFu) << 16));
    g1[3] = (int)(((dim1 >> 16) & 0xFFFFu) | (tile_k << 16));       // tile_dim0
    g1[4] = (int)(tile_rows & 0xFFFFu);                             // tile_dim1
    g1[5] = (int)dim0;                            // tensor_dim0_stride[31:0]
    g1[6] = 0;                                    // stride0 hi | stride1 lo
    g1[7] = 0;
    const i32x4 z4 = {0, 0, 0, 0};
    const i32x8 z8 = {0, 0, 0, 0, 0, 0, 0, 0};
    __builtin_amdgcn_tensor_load_to_lds(g0, g1, z4, z4, z8, 0);
}

// ---------------------------------------------------------------------------
// f32 -> bf16 (vectorized, grid-stride)
// ---------------------------------------------------------------------------
__global__ __launch_bounds__(256)
void cvt_f32_to_bf16(const float* __restrict__ x, unsigned short* __restrict__ y,
                     long long n4) {
    long long i = (long long)blockIdx.x * blockDim.x + threadIdx.x;
    long long stride = (long long)gridDim.x * blockDim.x;
    for (; i < n4; i += stride) {
        f32x4 v = ((const f32x4*)x)[i];
        u16x4 o;
        o[0] = f2bf(v[0]); o[1] = f2bf(v[1]); o[2] = f2bf(v[2]); o[3] = f2bf(v[3]);
        ((u16x4*)y)[i] = o;
    }
}

// W[K,N] f32 -> Wt[N,K] bf16
__global__ __launch_bounds__(256)
void transpose_cvt(const float* __restrict__ W, unsigned short* __restrict__ Wt,
                   int K, int N, long long total) {
    long long i = (long long)blockIdx.x * blockDim.x + threadIdx.x;
    long long stride = (long long)gridDim.x * blockDim.x;
    for (; i < total; i += stride) {
        long long n = i / K;
        long long k = i - n * (long long)K;
        Wt[i] = f2bf(W[k * (long long)N + n]);
    }
}

// ---------------------------------------------------------------------------
// GEMM: C[M,N] = A_bf[M,K] @ Bt_bf[N,K]^T   (bf16 in, f32 or bf16 out)
// Block = 4 waves covering 64(M) x 128(N); wave tile = 32 x 64 (8 WMMA C tiles).
// Per 32-wide K-step, wave 0 issues two TDM loads staging the block's A
// (64x32) and B (128x32) slices into double-buffered LDS; all waves read
// fragments from LDS (conflict-free b128 via 80-byte row pitch).
// A frag: lane row = l&15, elems i: K = (i/8)*16 + (l>>4)*8 + i%8
// B frag: lane col = l&15, elems i: K = (i/8)*16 + (l>>4)*8 + i%8
// ---------------------------------------------------------------------------
template <bool BF16OUT>
__global__ __launch_bounds__(128)
void gemm_bf16(const unsigned short* __restrict__ A,
               const unsigned short* __restrict__ Bt,
               void* __restrict__ C, int M, int N, int K) {
    extern __shared__ unsigned short lds[];
    // layout: A bufs [2][64][TPITCH] then B bufs [2][128][TPITCH]
    unsigned short* ldsA = lds;
    unsigned short* ldsB = lds + 2 * 64 * TPITCH;
    const unsigned A_BYTES = 64 * TPITCH * 2;     // 5120 B per buffer
    const unsigned B_BYTES = 128 * TPITCH * 2;    // 10240 B per buffer
    const unsigned B_BASE  = 2 * A_BYTES;         // byte offset of B region

    const int lane = threadIdx.x & 31;
    const int wave = threadIdx.x >> 5;
    const int wm = wave >> 1, wn = wave & 1;
    const int mblk = blockIdx.y * 64;
    const int nblk = blockIdx.x * 128;
    const int row = lane & 15, half = lane >> 4;

    v8f acc[2][4];
    const v8f vzero = {0.f, 0.f, 0.f, 0.f, 0.f, 0.f, 0.f, 0.f};
#pragma unroll
    for (int i = 0; i < 2; ++i)
#pragma unroll
        for (int j = 0; j < 4; ++j) acc[i][j] = vzero;

    const int steps = K >> 5;
    if (wave == 0) {
        tdm_load_2d(0,      A  + (size_t)mblk * K, 32, 64,  K);
        tdm_load_2d(B_BASE, Bt + (size_t)nblk * K, 32, 128, K);
    }

    for (int it = 0; it < steps; ++it) {
        const int cur = it & 1;
        if (wave == 0) __builtin_amdgcn_s_wait_tensorcnt(0);
        __syncthreads();   // publishes buf[cur]; all waves done with buf[1-cur]
        if (wave == 0 && it + 1 < steps) {
            const int nx = 1 - cur;
            const int k = (it + 1) * 32;
            tdm_load_2d(nx * A_BYTES,          A  + (size_t)mblk * K + k, 32, 64,  K);
            tdm_load_2d(B_BASE + nx * B_BYTES, Bt + (size_t)nblk * K + k, 32, 128, K);
        }

        const unsigned short* aBase = ldsA + cur * 64 * TPITCH;
        const unsigned short* bBase = ldsB + cur * 128 * TPITCH;

        Frag a0, a1;
        const unsigned short* a0p = aBase + (wm * 32 + row) * TPITCH;
        const unsigned short* a1p = aBase + (wm * 32 + 16 + row) * TPITCH;
        a0.q[0] = *(const u32x4*)(a0p + half * 8);
        a0.q[1] = *(const u32x4*)(a0p + 16 + half * 8);
        a1.q[0] = *(const u32x4*)(a1p + half * 8);
        a1.q[1] = *(const u32x4*)(a1p + 16 + half * 8);
#pragma unroll
        for (int t = 0; t < 4; ++t) {
            Frag b;
            const unsigned short* bp =
                bBase + (wn * 64 + t * 16 + row) * TPITCH + half * 8;
            b.q[0] = *(const u32x4*)(bp);
            b.q[1] = *(const u32x4*)(bp + 16);
            acc[0][t] = wmma_bf16(a0, b, acc[0][t]);
            acc[1][t] = wmma_bf16(a1, b, acc[1][t]);
        }
    }

#pragma unroll
    for (int mi = 0; mi < 2; ++mi)
#pragma unroll
        for (int t = 0; t < 4; ++t)
#pragma unroll
            for (int r = 0; r < 8; ++r) {
                int rr = mblk + wm * 32 + mi * 16 + 8 * half + r;
                int cc = nblk + wn * 64 + t * 16 + row;
                if (BF16OUT)
                    ((unsigned short*)C)[(size_t)rr * N + cc] = f2bf(acc[mi][t][r]);
                else
                    ((float*)C)[(size_t)rr * N + cc] = acc[mi][t][r];
            }
}

// ---------------------------------------------------------------------------
// RMSNorm + RoPE + repack. One wave per (token-row, 128-wide chunk).
// chunk 0..15: Q heads -> q_bf[B,NH,S,DH]   (scaled by 1/sqrt(DH))
// chunk 16..19: K heads -> k_bf[B,NKV,S,DH]
// chunk 20..23: V heads -> vT_bf[B,NKV,DH,S] (transposed copy)
// ---------------------------------------------------------------------------
__global__ __launch_bounds__(32)
void norm_rope_pack(const unsigned short* __restrict__ qkv,
                    const float* __restrict__ qw, const float* __restrict__ kw,
                    unsigned short* __restrict__ q_bf,
                    unsigned short* __restrict__ k_bf,
                    unsigned short* __restrict__ vT_bf) {
    const int chunk = blockIdx.x;        // 0..23
    const int rowi  = blockIdx.y;        // 0..4095
    const int b = rowi >> 11, s = rowi & (SEQ - 1);
    const int lane = threadIdx.x;

    const unsigned short* src = qkv + (size_t)rowi * QKV_N + chunk * DH + lane * 4;
    u16x4 raw = *(const u16x4*)src;

    if (chunk >= 20) {  // V: transpose copy, no norm/rope
        int hk = chunk - 20;
        unsigned short* dst =
            vT_bf + ((size_t)(b * NKV + hk) * DH + lane * 4) * SEQ + s;
        dst[0 * SEQ] = raw[0];
        dst[1 * SEQ] = raw[1];
        dst[2 * SEQ] = raw[2];
        dst[3 * SEQ] = raw[3];
        return;
    }

    const bool isq = chunk < 16;
    const float* w = isq ? qw : kw;

    float x0 = bf2f(raw[0]), x1 = bf2f(raw[1]), x2 = bf2f(raw[2]), x3 = bf2f(raw[3]);
    float ss = x0 * x0 + x1 * x1 + x2 * x2 + x3 * x3;
    ss = red_sum32(ss);
    float rms = rsqrtf(ss * (1.0f / (float)DH) + 1e-6f);

    float y[4] = { x0 * rms * w[lane * 4 + 0], x1 * rms * w[lane * 4 + 1],
                   x2 * rms * w[lane * 4 + 2], x3 * rms * w[lane * 4 + 3] };

    const float scale = isq ? 0.08838834764831845f /* 1/sqrt(128) */ : 1.0f;
    u16x4 o;
#pragma unroll
    for (int j = 0; j < 4; ++j) {
        int d = lane * 4 + j;
        int dm = d & 63;
        // freq = theta^(-dm/64) = exp(-dm * ln(10000)/64)
        float freq = __expf(-(float)dm * 0.14391156872f);
        float ang = (float)s * freq;
        float c = __cosf(ang), sn = __sinf(ang);
        float partner = __shfl_xor(y[j], 16);       // holds d +/- 64
        float rot = (d < 64) ? -partner : partner;
        o[j] = f2bf((y[j] * c + rot * sn) * scale);
    }

    unsigned short* dst;
    if (isq)
        dst = q_bf + ((size_t)(b * NH + chunk) * SEQ + s) * DH + lane * 4;
    else
        dst = k_bf + ((size_t)(b * NKV + (chunk - 16)) * SEQ + s) * DH + lane * 4;
    *(u16x4*)dst = o;
}

// ---------------------------------------------------------------------------
// Flash-style causal attention. One wave per (b, h, 16-query tile).
// Scores via 8 WMMAs per 32-key block, online softmax in C layout,
// probs bounced through LDS (C layout -> A layout), then 8 P@V WMMAs.
// Output written bf16 into attn[B*S, DMODEL] at column h*DH.
// ---------------------------------------------------------------------------
__global__ __launch_bounds__(32)
void attn_fwd(const unsigned short* __restrict__ q_bf,
              const unsigned short* __restrict__ k_bf,
              const unsigned short* __restrict__ vT_bf,
              unsigned short* __restrict__ attn_out) {
    __shared__ __align__(16) unsigned short pLDS[16 * 32];

    const int tile = blockIdx.x;   // 0..127
    const int h    = blockIdx.y;   // 0..15
    const int b    = blockIdx.z;   // 0..1
    const int hk   = h >> 2;
    const int lane = threadIdx.x;
    const int row = lane & 15, half = lane >> 4;

    const unsigned short* qp =
        q_bf + ((size_t)(b * NH + h) * SEQ + tile * 16) * DH;
    const unsigned short* kp = k_bf + (size_t)(b * NKV + hk) * SEQ * DH;
    const unsigned short* vp = vT_bf + (size_t)(b * NKV + hk) * DH * SEQ;

    Frag aq[4];                                   // Q tile, K-dim = 128
#pragma unroll
    for (int ks = 0; ks < 4; ++ks) {
        aq[ks].q[0] = *(const u32x4*)(qp + row * DH + ks * 32 + half * 8);
        aq[ks].q[1] = *(const u32x4*)(qp + row * DH + ks * 32 + 16 + half * 8);
    }

    const v8f vzero = {0.f, 0.f, 0.f, 0.f, 0.f, 0.f, 0.f, 0.f};
    v8f acc[8];
#pragma unroll
    for (int t = 0; t < 8; ++t) acc[t] = vzero;
    float m_i[8], l_i[8];
#pragma unroll
    for (int r = 0; r < 8; ++r) { m_i[r] = -3.0e38f; l_i[r] = 0.0f; }

    const int qrow_base = tile * 16 + 8 * half;
    const int nblk = ((tile * 16 + 15) >> 5) + 1;   // causal: key blocks of 32

    for (int kb = 0; kb < nblk; ++kb) {
        const int kbase = kb * 32;
        v8f s0 = vzero, s1 = vzero;
#pragma unroll
        for (int ks = 0; ks < 4; ++ks) {
            Frag b0, b1;
            const unsigned short* kp0 =
                kp + (size_t)(kbase + row) * DH + ks * 32 + half * 8;
            const unsigned short* kp1 =
                kp + (size_t)(kbase + 16 + row) * DH + ks * 32 + half * 8;
            b0.q[0] = *(const u32x4*)(kp0);
            b0.q[1] = *(const u32x4*)(kp0 + 16);
            b1.q[0] = *(const u32x4*)(kp1);
            b1.q[1] = *(const u32x4*)(kp1 + 16);
            s0 = wmma_bf16(aq[ks], b0, s0);
            s1 = wmma_bf16(aq[ks], b1, s1);
        }

        const int k0 = kbase + row, k1 = kbase + 16 + row;
#pragma unroll
        for (int r = 0; r < 8; ++r) {
            const int qg = qrow_base + r;
            float v0 = (k0 <= qg) ? s0[r] : -3.0e38f;   // causal mask
            float v1 = (k1 <= qg) ? s1[r] : -3.0e38f;
            float mblk = red_max16(fmaxf(v0, v1));
            float mnew = fmaxf(m_i[r], mblk);
            float sc = __expf(m_i[r] - mnew);
            m_i[r] = mnew;
            l_i[r] *= sc;
#pragma unroll
            for (int t = 0; t < 8; ++t) acc[t][r] *= sc;
            float p0 = __expf(v0 - mnew);
            float p1 = __expf(v1 - mnew);
            l_i[r] += red_sum16(p0 + p1);
            // scatter probs (C layout) into LDS as [16 q][32 k]
            pLDS[(8 * half + r) * 32 + row] = f2bf(p0);
            pLDS[(8 * half + r) * 32 + 16 + row] = f2bf(p1);
        }
        __syncthreads();

        Frag pa;   // re-load probs in A layout
        pa.q[0] = *(const u32x4*)&pLDS[row * 32 + half * 8];
        pa.q[1] = *(const u32x4*)&pLDS[row * 32 + 16 + half * 8];
#pragma unroll
        for (int t = 0; t < 8; ++t) {
            Frag bv;
            const unsigned short* vpp =
                vp + (size_t)(t * 16 + row) * SEQ + kbase + half * 8;
            bv.q[0] = *(const u32x4*)(vpp);
            bv.q[1] = *(const u32x4*)(vpp + 16);
            acc[t] = wmma_bf16(pa, bv, acc[t]);
        }
        __syncthreads();
    }

#pragma unroll
    for (int r = 0; r < 8; ++r) {
        const int qg = qrow_base + r;
        const float inv = 1.0f / l_i[r];
        const size_t rowoff = ((size_t)b * SEQ + qg) * DMODEL + h * DH;
#pragma unroll
        for (int t = 0; t < 8; ++t)
            attn_out[rowoff + t * 16 + row] = f2bf(acc[t][r] * inv);
    }
}

// ---------------------------------------------------------------------------
// Host side: workspace layout (100 MiB total) + launches
// ---------------------------------------------------------------------------
extern "C" void kernel_launch(void* const* d_in, const int* in_sizes, int n_in,
                              void* d_out, int out_size, void* d_ws, size_t ws_size,
                              hipStream_t stream) {
    (void)in_sizes; (void)n_in; (void)out_size; (void)ws_size;
    const float* x  = (const float*)d_in[0];
    const float* Wq = (const float*)d_in[1];
    const float* Wk = (const float*)d_in[2];
    const float* Wv = (const float*)d_in[3];
    const float* Wo = (const float*)d_in[4];
    const float* qw = (const float*)d_in[5];
    const float* kw = (const float*)d_in[6];

    char* ws = (char*)d_ws;
    unsigned short* xb     = (unsigned short*)(ws + 0);          // 16 MiB  x bf16 [4096,2048]
    unsigned short* wqkvT  = (unsigned short*)(ws + 16777216);   // 12 MiB  [3072,2048]
    unsigned short* woT    = (unsigned short*)(ws + 29360128);   // 8 MiB   [2048,2048]
    unsigned short* qkvbf  = (unsigned short*)(ws + 37748736);   // 24 MiB  [4096,3072]
    unsigned short* qbf    = (unsigned short*)(ws + 62914560);   // 16 MiB  [B,NH,S,DH]
    unsigned short* kbf    = (unsigned short*)(ws + 79691776);   // 4 MiB   [B,NKV,S,DH]
    unsigned short* vtbf   = (unsigned short*)(ws + 83886080);   // 4 MiB   [B,NKV,DH,S]
    unsigned short* attnbf = (unsigned short*)(ws + 88080384);   // 16 MiB  [4096,2048]

    const unsigned gemm_lds = 2 * (64 * TPITCH * 2) + 2 * (128 * TPITCH * 2); // 30720 B

    // 1) precision demotion / weight transposes
    cvt_f32_to_bf16<<<2048, 256, 0, stream>>>(x, xb, (long long)ROWS * DMODEL / 4);
    transpose_cvt<<<4096, 256, 0, stream>>>(Wq, wqkvT, DMODEL, DMODEL,
                                            (long long)DMODEL * DMODEL);
    transpose_cvt<<<1024, 256, 0, stream>>>(Wk, wqkvT + (size_t)DMODEL * DMODEL,
                                            DMODEL, NKV * DH,
                                            (long long)NKV * DH * DMODEL);
    transpose_cvt<<<1024, 256, 0, stream>>>(Wv, wqkvT + (size_t)(DMODEL + NKV * DH) * DMODEL,
                                            DMODEL, NKV * DH,
                                            (long long)NKV * DH * DMODEL);
    transpose_cvt<<<4096, 256, 0, stream>>>(Wo, woT, DMODEL, DMODEL,
                                            (long long)DMODEL * DMODEL);

    // 2) fused QKV projection (bf16 out), TDM-staged LDS double buffering
    gemm_bf16<true><<<dim3(QKV_N / 128, ROWS / 64), 128, gemm_lds, stream>>>(
        xb, wqkvT, qkvbf, ROWS, QKV_N, DMODEL);

    // 3) RMSNorm + RoPE + repack (V transposed)
    norm_rope_pack<<<dim3(QKV_N / DH, ROWS), 32, 0, stream>>>(
        qkvbf, qw, kw, qbf, kbf, vtbf);

    // 4) causal flash attention
    attn_fwd<<<dim3(SEQ / 16, NH, BATCH), 32, 0, stream>>>(qbf, kbf, vtbf, attnbf);

    // 5) output projection (f32 out), TDM-staged LDS double buffering
    gemm_bf16<false><<<dim3(DMODEL / 128, ROWS / 64), 128, gemm_lds, stream>>>(
        attnbf, woT, d_out, ROWS, DMODEL, DMODEL);
}
